// seqFusionAttentionWeightProj_48782238548116
// MI455X (gfx1250) — matrix operand order repaired and checked by reading, not compiled, not run on hardware
//
#include <hip/hip_runtime.h>
#include <hip/hip_bf16.h>

// ---------------------------------------------------------------------------
// Problem constants (match reference)
// ---------------------------------------------------------------------------
constexpr int Bsz  = 2;
constexpr int Sq   = 2048;
constexpr int Hh   = 8;     // heads
constexpr int Cc   = 32;    // head dim
constexpr int WIN  = 32;
constexpr int NROWS = Bsz * Sq;   // 4096

typedef __attribute__((ext_vector_type(16))) __bf16 v16bf;
typedef __attribute__((ext_vector_type(8)))  float  v8f;
typedef __attribute__((ext_vector_type(4)))  unsigned int u32x4;
typedef __attribute__((ext_vector_type(8)))  int i32x8;
typedef __attribute__((ext_vector_type(4)))  int i32x4;

union FragAB {
  v16bf v;
  unsigned short u[16];
  uint4 q[2];
};

__device__ __forceinline__ unsigned short f2bf(float x) {
  unsigned int u = __float_as_uint(x);
  u += 0x7FFFu + ((u >> 16) & 1u);   // round-to-nearest-even
  return (unsigned short)(u >> 16);
}

// ---------------------------------------------------------------------------
// Tensor Data Mover: DMA one 16x256 tile (row stride 256 elements) from global
// memory into LDS. elem_code: 1 = 2-byte elements (bf16), 2 = 4-byte (fp32).
// D# layout per CDNA5 ISA ch.8 (group0: count/lds/global/type, group1: dims).
// Issue from ONE wave; caller must s_wait_tensorcnt + workgroup barrier.
// ---------------------------------------------------------------------------
__device__ __forceinline__ void tdm_load_16x256(unsigned int lds_off,
                                                const void* gptr,
                                                unsigned int elem_code) {
  const unsigned long long ga = (unsigned long long)(uintptr_t)gptr;
  u32x4 g0;
  g0[0] = 1u;                                            // count=1, user descriptor
  g0[1] = lds_off;                                       // lds_addr (bytes)
  g0[2] = (unsigned int)(ga & 0xFFFFFFFFull);            // global_addr[31:0]
  g0[3] = (unsigned int)((ga >> 32) & 0x01FFFFFFull)     // global_addr[56:32]
          | (2u << 30);                                  // type=2 ("image")
  i32x8 g1;
  g1[0] = (int)(elem_code << 16);                        // wg_mask=0, data_size
  g1[1] = (int)(256u << 16);                             // tensor_dim0[15:0]=256
  g1[2] = (int)(16u << 16);                              // dim0 hi=0 | tensor_dim1[15:0]=16
  g1[3] = (int)(256u << 16);                             // dim1 hi=0 | tile_dim0=256
  g1[4] = 16;                                            // tile_dim1=16, tile_dim2=0
  g1[5] = 256;                                           // tensor_dim0_stride[31:0]
  g1[6] = 0;                                             // stride hi | dim1_stride lo
  g1[7] = 0;
  i32x4 g2 = {0, 0, 0, 0};                               // 2D tensor: groups 2/3 unused
  i32x4 g3 = {0, 0, 0, 0};
#if __clang_major__ >= 23
  i32x8 g4 = {0, 0, 0, 0, 0, 0, 0, 0};
  __builtin_amdgcn_tensor_load_to_lds(g0, g1, g2, g3, g4, 0);
#else
  __builtin_amdgcn_tensor_load_to_lds(g0, g1, g2, g3, 0);
#endif
}

// ---------------------------------------------------------------------------
// Weight conversion: fp32 -> bf16, laid out [outcol][k] (k contiguous) so
// B-fragments are contiguous 32B loads per lane.
// QTrans/KTrans/VTrans come in as [k][o] -> transpose. GW/OW already [o][k].
// ---------------------------------------------------------------------------
__global__ __launch_bounds__(256)
void convert_weights(const float* __restrict__ QT, const float* __restrict__ KT,
                     const float* __restrict__ VT, const float* __restrict__ GW,
                     const float* __restrict__ OW,
                     unsigned short* __restrict__ WQ, unsigned short* __restrict__ WK,
                     unsigned short* __restrict__ WV, unsigned short* __restrict__ WG,
                     unsigned short* __restrict__ WO) {
  int i = blockIdx.x * 256 + threadIdx.x;
  if (i >= 256 * 256) return;
  int o = i >> 8, k = i & 255;
  WQ[i] = f2bf(QT[k * 256 + o]);
  WK[i] = f2bf(KT[k * 256 + o]);
  WV[i] = f2bf(VT[k * 256 + o]);
  WG[i] = f2bf(GW[i]);
  WO[i] = f2bf(OW[i]);
}

// ---------------------------------------------------------------------------
// Generic 16-row-slab GEMM: P[16,256] = X[16,256] * W^T  (+bias/activation)
// Block = 128 threads = 4 waves; wave w owns column tiles 4w..4w+3.
// Input tile staged to LDS by the Tensor Data Mover (wave 0 issues, waits
// TENSORcnt, then workgroup barrier). fp32 inputs are converted in LDS.
// K accumulated in 8 steps of 32 with v_wmma_f32_16x16x32_bf16.
// ---------------------------------------------------------------------------
#define GEMM_OUT_BF16    0   // store bf16 row-major (+ optional V^T per-head store)
#define GEMM_SIGMOID_F32 1   // store f32 sigmoid(acc + bias)
#define GEMM_BIAS_F32    2   // store f32 acc + bias

__global__ __launch_bounds__(128)
void gemm16_bf16wmma(const float* __restrict__ Xf,            // fp32 input (or null)
                     const unsigned short* __restrict__ Xb,   // bf16 input (or null)
                     const unsigned short* __restrict__ W,    // [256 out][256 k] bf16
                     const float* __restrict__ bias,          // [256] or null
                     unsigned short* __restrict__ outB,       // bf16 [N][256] (mode 0)
                     float* __restrict__ outF,                // f32  [N][256] (mode 1/2)
                     unsigned short* __restrict__ vt,         // bf16 [B][H][32][S] (or null)
                     int mode) {
  extern __shared__ char smem[];
  unsigned short* ldsA = (unsigned short*)smem;            // 8 KB bf16 A tile
  float*          ldsF = (float*)(smem + 16 * 256 * 2);    // 16 KB fp32 staging (fp32 path)

  const int tid  = threadIdx.x;
  const int row0 = blockIdx.x * 16;

  // --- TDM stage: wave 0 DMAs the input tile into LDS ---
  if (tid < 32) {
    if (Xf) {
      tdm_load_16x256((unsigned int)(uintptr_t)ldsF, Xf + (size_t)row0 * 256, 2u);
    } else {
      tdm_load_16x256((unsigned int)(uintptr_t)ldsA, Xb + (size_t)row0 * 256, 1u);
    }
    __builtin_amdgcn_s_wait_tensorcnt(0);
  }
  __syncthreads();

  if (Xf) {  // convert staged fp32 tile -> bf16 tile in LDS
#pragma unroll
    for (int i = tid; i < 16 * 256; i += 128) ldsA[i] = f2bf(ldsF[i]);
    __syncthreads();
  }

  const int wave = tid >> 5;
  const int lane = tid & 31;
  const int hlf  = lane >> 4;
  const int nm   = lane & 15;   // M for A-frag, N for B/C-frag

#pragma unroll
  for (int cti = 0; cti < 4; ++cti) {
    const int ct   = wave * 4 + cti;
    const int ocol = ct * 16 + nm;                       // this lane's output column
    const unsigned short* wrow = W + (size_t)ocol * 256; // contiguous k
    __builtin_prefetch(wrow + 16 * 256, 0, 3);           // next col-tile's row -> global_prefetch

    v8f acc = {};
#pragma unroll
    for (int k = 0; k < 8; ++k) {
      const int kb = k * 32;
      FragAB a, b;
      // A 16x32 bf16 layout: lane holds M=lane%16; elems 0..7 -> K=kb+hlf*8+j,
      // elems 8..15 -> K=kb+16+hlf*8+j
      a.q[0] = *(const uint4*)(ldsA + nm * 256 + kb + hlf * 8);
      a.q[1] = *(const uint4*)(ldsA + nm * 256 + kb + 16 + hlf * 8);
      // B 32x16 bf16 layout: lane holds N; same K order as A
      b.q[0] = *(const uint4*)(wrow + kb + hlf * 8);
      b.q[1] = *(const uint4*)(wrow + kb + 16 + hlf * 8);
      acc = __builtin_amdgcn_wmma_f32_16x16x32_bf16(false, a.v, false, b.v,
                                                    (short)0, acc, false, false);
    }

    const float bv = bias ? bias[ocol] : 0.0f;
    if (mode == GEMM_OUT_BF16) {
      unsigned short t8[8];
#pragma unroll
      for (int r = 0; r < 8; ++r) {   // C/D frag: VGPR r -> row M = hlf*8 + r
        const int rowabs = row0 + hlf * 8 + r;
        unsigned short hv = f2bf(acc[r]);
        outB[(size_t)rowabs * 256 + ocol] = hv;
        t8[r] = hv;
      }
      if (vt) {  // also emit V^T per head: Vt[b][h][c][t], t contiguous per lane
        const int rowa = row0 + hlf * 8;
        const int bidx = rowa / Sq, s = rowa % Sq;
        const int hh = ocol >> 5, cc = ocol & 31;
        uint4 pack;
        unsigned int* pu = (unsigned int*)&pack;
#pragma unroll
        for (int r = 0; r < 4; ++r)
          pu[r] = (unsigned int)t8[2 * r] | ((unsigned int)t8[2 * r + 1] << 16);
        *(uint4*)(vt + (((size_t)bidx * Hh + hh) * Cc + cc) * Sq + s) = pack;
      }
    } else {
#pragma unroll
      for (int r = 0; r < 8; ++r) {
        const int rowabs = row0 + hlf * 8 + r;
        float v = acc[r] + bv;
        if (mode == GEMM_SIGMOID_F32) v = 1.0f / (1.0f + __expf(-v));
        outF[(size_t)rowabs * 256 + ocol] = v;
      }
    }
  }
}

// ---------------------------------------------------------------------------
// Local-window attention, one wave per (16 queries, head, batch).
// Scores: 5 WMMAs (K tiles s0-32 .. s0+48), masked softmax via shfl_xor
// reductions in 16-lane groups, probs -> LDS (padded to K=96), A*V: 2x3 WMMAs,
// gated, stored bf16 for the final projection GEMM.
// ---------------------------------------------------------------------------
__global__ __launch_bounds__(32)
void local_attn_wmma(const unsigned short* __restrict__ Qp,
                     const unsigned short* __restrict__ Kp,
                     const unsigned short* __restrict__ Vt,   // [B][H][32][S]
                     const float* __restrict__ Gp,            // f32 [N][256]
                     unsigned short* __restrict__ AttO) {     // bf16 [N][256]
  __shared__ unsigned short probs[16 * 96];   // 3 KB, zero padded cols 80..95
  const int lane = threadIdx.x;
  const int hlf  = lane >> 4;
  const int nm   = lane & 15;
  const int s0   = blockIdx.x * 16;
  const int h    = blockIdx.y;
  const int b    = blockIdx.z;
  const size_t rowbase = (size_t)b * Sq;

#pragma unroll
  for (int i = lane; i < 16 * 96 / 2; i += 32) ((unsigned int*)probs)[i] = 0u;

  // Q fragment (A 16x32, K-dim == C == 32 exactly)
  FragAB qf;
  {
    const unsigned short* qp = Qp + (rowbase + s0 + nm) * 256 + h * 32;
    qf.q[0] = *(const uint4*)(qp + hlf * 8);
    qf.q[1] = *(const uint4*)(qp + 16 + hlf * 8);
  }

  // Scores against the 5 in-window key tiles
  v8f sc[5];
#pragma unroll
  for (int tt = 0; tt < 5; ++tt) {
    int t = s0 - 32 + tt * 16 + nm;                 // this lane's key position
    t = t < 0 ? 0 : (t >= Sq ? Sq - 1 : t);         // clamp; mask kills OOB below
    const unsigned short* kp = Kp + (rowbase + t) * 256 + h * 32;
    FragAB kf;
    kf.q[0] = *(const uint4*)(kp + hlf * 8);
    kf.q[1] = *(const uint4*)(kp + 16 + hlf * 8);
    v8f z = {};
    sc[tt] = __builtin_amdgcn_wmma_f32_16x16x32_bf16(false, qf.v, false, kf.v,
                                                     (short)0, z, false, false);
  }

  const float scale = 0.17677669529663687f;  // 1/sqrt(C)
#pragma unroll
  for (int r = 0; r < 8; ++r) {
    const int sabs = s0 + hlf * 8 + r;       // row owned by this VGPR slot
    float vals[5];
    float mx = -3.0e38f;
#pragma unroll
    for (int tt = 0; tt < 5; ++tt) {
      const int t = s0 - 32 + tt * 16 + nm;
      const int d = sabs - t;
      const bool ok = (t >= 0) && (t < Sq) && (d <= WIN) && (d >= -WIN);
      float v = ok ? sc[tt][r] * scale : -1.0e30f;
      vals[tt] = v;
      mx = fmaxf(mx, v);
    }
#pragma unroll
    for (int m = 1; m < 16; m <<= 1) mx = fmaxf(mx, __shfl_xor(mx, m, 32));
    float sum = 0.0f;
#pragma unroll
    for (int tt = 0; tt < 5; ++tt) { vals[tt] = __expf(vals[tt] - mx); sum += vals[tt]; }
#pragma unroll
    for (int m = 1; m < 16; m <<= 1) sum += __shfl_xor(sum, m, 32);
    const float inv = 1.0f / sum;
    const int mr = hlf * 8 + r;
#pragma unroll
    for (int tt = 0; tt < 5; ++tt)
      probs[mr * 96 + tt * 16 + nm] = f2bf(vals[tt] * inv);
  }
  __syncthreads();

  // O[16,32] = P[16,96] * V[96,32]   (zero-padded tail), then gate + store
#pragma unroll
  for (int nt = 0; nt < 2; ++nt) {
    const int c = nt * 16 + nm;
    const unsigned short* vrow = Vt + (((size_t)b * Hh + h) * Cc + c) * Sq;
    v8f acc = {};
#pragma unroll
    for (int ks = 0; ks < 3; ++ks) {
      FragAB a, vb;
      a.q[0] = *(const uint4*)(probs + nm * 96 + ks * 32 + hlf * 16);
      a.q[1] = *(const uint4*)(probs + nm * 96 + ks * 32 + hlf * 16 + 8);
      int tseg = s0 - 32 + ks * 32 + hlf * 16;
      tseg = tseg < 0 ? 0 : (tseg > Sq - 16 ? Sq - 16 : tseg);  // zero-prob regions
      vb.q[0] = *(const uint4*)(vrow + tseg);
      vb.q[1] = *(const uint4*)(vrow + tseg + 8);
      acc = __builtin_amdgcn_wmma_f32_16x16x32_bf16(false, a.v, false, vb.v,
                                                    (short)0, acc, false, false);
    }
#pragma unroll
    for (int r = 0; r < 8; ++r) {
      const size_t row = rowbase + s0 + hlf * 8 + r;
      const int col = h * 32 + nt * 16 + nm;
      const float g = Gp[row * 256 + col];
      AttO[row * 256 + col] = f2bf(acc[r] * g);
    }
  }
}

// ---------------------------------------------------------------------------
// Launch
// ---------------------------------------------------------------------------
extern "C" void kernel_launch(void* const* d_in, const int* in_sizes, int n_in,
                              void* d_out, int out_size, void* d_ws, size_t ws_size,
                              hipStream_t stream) {
  const float* Qin  = (const float*)d_in[0];
  const float* Kin  = (const float*)d_in[1];
  const float* Vin  = (const float*)d_in[2];
  const float* QT   = (const float*)d_in[3];
  const float* KT   = (const float*)d_in[4];
  const float* VT   = (const float*)d_in[5];
  const float* GW   = (const float*)d_in[6];
  const float* Gb   = (const float*)d_in[7];
  const float* OW   = (const float*)d_in[8];
  const float* Ob   = (const float*)d_in[9];
  float* out = (float*)d_out;

  char* ws = (char*)d_ws;
  size_t off = 0;
  auto take = [&](size_t bytes) -> void* {
    void* p = ws + off;
    off = (off + bytes + 255) & ~(size_t)255;
    return p;
  };
  unsigned short* WQ = (unsigned short*)take(256 * 256 * 2);
  unsigned short* WK = (unsigned short*)take(256 * 256 * 2);
  unsigned short* WV = (unsigned short*)take(256 * 256 * 2);
  unsigned short* WG = (unsigned short*)take(256 * 256 * 2);
  unsigned short* WO = (unsigned short*)take(256 * 256 * 2);
  unsigned short* Qp = (unsigned short*)take((size_t)NROWS * 256 * 2);
  unsigned short* Kp = (unsigned short*)take((size_t)NROWS * 256 * 2);
  unsigned short* Vp = (unsigned short*)take((size_t)NROWS * 256 * 2);  // row-major V (scratch)
  unsigned short* Vt = (unsigned short*)take((size_t)NROWS * 256 * 2);  // [B][H][32][S]
  float*          Gp = (float*)take((size_t)NROWS * 256 * 4);
  unsigned short* AttO = (unsigned short*)take((size_t)NROWS * 256 * 2);

  convert_weights<<<256, 256, 0, stream>>>(QT, KT, VT, GW, OW, WQ, WK, WV, WG, WO);

  const int gb = NROWS / 16;                  // 256 blocks
  const size_t lds_f32  = 16 * 256 * 2 + 16 * 256 * 4;  // bf16 tile + fp32 staging
  const size_t lds_bf16 = 16 * 256 * 2 + 16;            // bf16 tile only (+pad)

  gemm16_bf16wmma<<<gb, 128, lds_f32, stream>>>(Qin, nullptr, WQ, nullptr, Qp, nullptr, nullptr, GEMM_OUT_BF16);
  gemm16_bf16wmma<<<gb, 128, lds_f32, stream>>>(Kin, nullptr, WK, nullptr, Kp, nullptr, nullptr, GEMM_OUT_BF16);
  gemm16_bf16wmma<<<gb, 128, lds_f32, stream>>>(Vin, nullptr, WV, nullptr, Vp, nullptr, Vt,      GEMM_OUT_BF16);
  gemm16_bf16wmma<<<gb, 128, lds_f32, stream>>>(Vin, nullptr, WG, Gb,      nullptr, Gp, nullptr, GEMM_SIGMOID_F32);

  dim3 ag(Sq / 16, Hh, Bsz);  // 128 x 8 x 2
  local_attn_wmma<<<ag, 32, 0, stream>>>(Qp, Kp, Vt, Gp, AttO);

  gemm16_bf16wmma<<<gb, 128, lds_bf16, stream>>>(nullptr, AttO, WO, Ob, nullptr, out, nullptr, GEMM_BIAS_F32);
}
